// LSTM_Sequential_State_41841571397843
// MI455X (gfx1250) — compile-verified, matrix operand order checked by reading, not run on hardware
//
#include <hip/hip_runtime.h>
#include <stdint.h>

// LSTM fused kernel for MI455X (gfx1250, wave32, WMMA).
// B=256, T=2048, D=64, U=64. One workgroup per 16-row batch tile (16 WGs),
// 4 waves per WG, each wave owns one 16-wide u-column block and computes all
// four gates for it -> gate math fully register-resident, 1 barrier/step.

typedef __attribute__((ext_vector_type(16))) __bf16        v16bf;
typedef __attribute__((ext_vector_type(2)))  __bf16        v2bf;
typedef __attribute__((ext_vector_type(8)))  float         v8f;
typedef __attribute__((ext_vector_type(4)))  unsigned int  u32x4;
typedef __attribute__((ext_vector_type(4)))  float         f32x4;

#define BQ 256
#define TQ 2048
#define DQ 64
#define UQ 64
#define G4 256            // 4*U
#define ROWSTRIDE 136     // 128 bf16 elems + 8 pad (272B row -> 4-bank skew)

__device__ __forceinline__ uint32_t pack2_bf16(float a, float b) {
#if __has_builtin(__builtin_amdgcn_cvt_pk_bf16_f32)
  union { v2bf v; uint32_t u; } cv;
  cv.v = __builtin_amdgcn_cvt_pk_bf16_f32(a, b);     // v_cvt_pk_bf16_f32
  return cv.u;
#else
  uint32_t ua = __float_as_uint(a); ua += 0x7FFFu + ((ua >> 16) & 1u);
  uint32_t ub = __float_as_uint(b); ub += 0x7FFFu + ((ub >> 16) & 1u);
  return (ua >> 16) | (ub & 0xFFFF0000u);
#endif
}
__device__ __forceinline__ uint32_t bf16_bits(float a) {
  return pack2_bf16(a, 0.0f) & 0xFFFFu;
}

__device__ __forceinline__ float ftanh(float x) {
#if __has_builtin(__builtin_amdgcn_tanhf)
  return __builtin_amdgcn_tanhf(x);                  // v_tanh_f32 (1 TRANS op)
#else
  float e = __builtin_amdgcn_exp2f(x * -2.88539008f);  // 2^(-2x*log2e)
  return 2.0f * __builtin_amdgcn_rcpf(1.0f + e) - 1.0f;
#endif
}
__device__ __forceinline__ float fsigmoid(float x) {
#if __has_builtin(__builtin_amdgcn_tanhf)
  // sigmoid(x) = 0.5*tanh(x/2) + 0.5  -> v_mul + v_tanh_f32 + v_fma
  return __builtin_fmaf(0.5f, __builtin_amdgcn_tanhf(0.5f * x), 0.5f);
#else
  float e = __builtin_amdgcn_exp2f(x * -1.44269504f);
  return __builtin_amdgcn_rcpf(1.0f + e);
#endif
}

union FragU { v16bf v; u32x4 q[2]; uint32_t u[8]; };

__global__ __launch_bounds__(128, 1)
void lstm_fused_wmma(const float* __restrict__ x,
                     const float* __restrict__ Wx,
                     const float* __restrict__ Wh,
                     const float* __restrict__ b,
                     float* __restrict__ out)
{
  // Double-buffered concatenated activation tile [x_t | h_t] : 16 x 128 bf16.
  __shared__ uint16_t Abuf[2][16][ROWSTRIDE];

  const int tid  = threadIdx.x;
  const int lane = tid & 31;
  const int wv   = tid >> 5;          // wave 0..3 -> u-block
  const int b0   = blockIdx.x * 16;   // batch tile base row
  const int n    = lane & 15;         // tile column (B/C/D frag) or A row
  const int half = lane >> 4;         // lane group 0/1
  const int u0   = wv * 16;

  // ---- Load weight (B) fragments once: combined W = [Wx ; Wh], K=128 ----
  // 16x16x32 bf16 B-frag lane layout (mirror of A): lanes 0-15 hold K 0..7
  // and 16..23, lanes 16-31 hold K 8..15 and 24..31 (pairs per VGPR).
  FragU fragB[4][4];                  // [gate][k-chunk]
  #pragma unroll
  for (int g = 0; g < 4; ++g) {
    const int col = g * UQ + u0 + n;  // column in 4U gate dimension
    #pragma unroll
    for (int kc = 0; kc < 4; ++kc) {
      #pragma unroll
      for (int v = 0; v < 8; ++v) {
        const int kl = (v < 4 ? 2 * v : 16 + 2 * (v - 4)) + half * 8;
        const int k0 = kc * 32 + kl;
        const float* r0 = (k0     < DQ) ? (Wx + (size_t)k0 * G4)
                                        : (Wh + (size_t)(k0 - DQ) * G4);
        const float* r1 = (k0 + 1 < DQ) ? (Wx + (size_t)(k0 + 1) * G4)
                                        : (Wh + (size_t)(k0 + 1 - DQ) * G4);
        fragB[g][kc].u[v] = pack2_bf16(r0[col], r1[col]);
      }
    }
  }
  float biasv[4];
  #pragma unroll
  for (int g = 0; g < 4; ++g) biasv[g] = b[g * UQ + u0 + n];

  // ---- Init: zero h-half of buffer 0, stage x_0 into x-half ----
  {
    const int r  = tid >> 3;          // 0..15 rows
    const int cg = (tid & 7) * 8;     // 0..56 col group (8 elems)
    u32x4 z4 = {0u, 0u, 0u, 0u};
    *reinterpret_cast<u32x4*>(&Abuf[0][r][DQ + cg]) = z4;       // h = 0
    const float* xp = x + ((size_t)(b0 + r) * TQ + 0) * DQ + cg;
    f32x4 a0 = *reinterpret_cast<const f32x4*>(xp);
    f32x4 a1 = *reinterpret_cast<const f32x4*>(xp + 4);
    u32x4 p;
    p.x = pack2_bf16(a0.x, a0.y); p.y = pack2_bf16(a0.z, a0.w);
    p.z = pack2_bf16(a1.x, a1.y); p.w = pack2_bf16(a1.z, a1.w);
    *reinterpret_cast<u32x4*>(&Abuf[0][r][cg]) = p;             // x_0
  }
  float cst[8];
  #pragma unroll
  for (int r = 0; r < 8; ++r) cst[r] = 0.0f;
  __syncthreads();

  // ---- Sequential scan over T ----
  for (int t = 0; t < TQ; ++t) {
    const int cb = t & 1, nb = cb ^ 1;

    // A-fragments [x_t | h_t] from LDS: 2x b128 per 32-K chunk per lane.
    FragU fragA[4];
    #pragma unroll
    for (int kc = 0; kc < 4; ++kc) {
      const int c0 = kc * 32 + half * 8;
      fragA[kc].q[0] = *reinterpret_cast<const u32x4*>(&Abuf[cb][n][c0]);
      fragA[kc].q[1] = *reinterpret_cast<const u32x4*>(&Abuf[cb][n][c0 + 16]);
    }

    // Stage x_{t+1}: issue global loads early to overlap with WMMA.
    u32x4 xpk = {0u, 0u, 0u, 0u};
    const int  xr = tid >> 3, xc = (tid & 7) * 8;
    const bool do_stage = (t + 1 < TQ);
    if (do_stage) {
      const float* xp = x + ((size_t)(b0 + xr) * TQ + (t + 1)) * DQ + xc;
      f32x4 a0 = *reinterpret_cast<const f32x4*>(xp);
      f32x4 a1 = *reinterpret_cast<const f32x4*>(xp + 4);
      __builtin_prefetch(xp + DQ, 0, 3);          // x_{t+2}, near-temporal
      xpk.x = pack2_bf16(a0.x, a0.y); xpk.y = pack2_bf16(a0.z, a0.w);
      xpk.z = pack2_bf16(a1.x, a1.y); xpk.w = pack2_bf16(a1.z, a1.w);
    }

    // z = [x_t|h_t] @ [Wx;Wh] + b : 4 gates x 4 K-chunks = 16 bf16 WMMAs.
    v8f zacc[4];
    #pragma unroll
    for (int g = 0; g < 4; ++g) {
      v8f acc;
      #pragma unroll
      for (int r = 0; r < 8; ++r) acc[r] = biasv[g];
      #pragma unroll
      for (int kc = 0; kc < 4; ++kc) {
        acc = __builtin_amdgcn_wmma_f32_16x16x32_bf16(
            false, fragA[kc].v, false, fragB[g][kc].v,
            (short)0, acc, false, false);
      }
      zacc[g] = acc;
    }

    // Gate math fully in registers; C/D layout: lane n holds column u0+n,
    // VGPR r holds row half*8 + r.
    const size_t orow0 = ((size_t)(b0 + half * 8) * TQ + t) * (2 * UQ) + u0 + n;
    #pragma unroll
    for (int r = 0; r < 8; ++r) {
      const float iv = fsigmoid(zacc[0][r]);
      const float fv = fsigmoid(zacc[1][r]);
      const float gv = ftanh   (zacc[2][r]);
      const float ov = fsigmoid(zacc[3][r]);
      const float cn = fv * cst[r] + iv * gv;
      cst[r] = cn;
      const float hn = ov * ftanh(cn);
      const size_t oidx = orow0 + (size_t)r * ((size_t)TQ * 2 * UQ);
      out[oidx]      = cn;                         // cell state  [.., 0:64]
      out[oidx + UQ] = hn;                         // hidden      [..,64:128]
      Abuf[nb][half * 8 + r][DQ + u0 + n] = (uint16_t)bf16_bits(hn);
    }

    if (do_stage)
      *reinterpret_cast<u32x4*>(&Abuf[nb][xr][xc]) = xpk;

    __syncthreads();   // next buffer fully written; current buffer reads done
  }
}

extern "C" void kernel_launch(void* const* d_in, const int* in_sizes, int n_in,
                              void* d_out, int out_size, void* d_ws, size_t ws_size,
                              hipStream_t stream) {
  (void)in_sizes; (void)n_in; (void)out_size; (void)d_ws; (void)ws_size;
  const float* x  = (const float*)d_in[0];   // [B,T,D]
  const float* Wx = (const float*)d_in[1];   // [D,4U]
  const float* Wh = (const float*)d_in[2];   // [U,4U]
  const float* b  = (const float*)d_in[3];   // [4U]
  float* out = (float*)d_out;                // [B,T,2U]
  lstm_fused_wmma<<<dim3(BQ / 16), dim3(128), 0, stream>>>(x, Wx, Wh, b, out);
}
